// DSNN_3058016714808
// MI455X (gfx1250) — matrix operand order repaired and checked by reading, not compiled, not run on hardware
//
#include <hip/hip_runtime.h>
#include <math.h>

// ---------------------------------------------------------------------------
// Spiking-net restructured for MI455X (gfx1250):
//  - h0 = x@W0 hoisted out of the time loop (time-invariant input)
//  - layer-0 recurrence is elementwise -> packed spike bitmasks (L2-resident)
//  - layer-1: 127 x [16384x512]@[512x512] f16 WMMA; A synthesized from
//    bitmasks in VALU; syn scaled by ALPHA then WMMA accumulates h directly
//    into it (C-operand fusion); 2 M-tiles per wave reuse each B fragment 2x
//  - layer-2 collapses linearly (no reset) to ONE GEMM: out = z @ W2
// ---------------------------------------------------------------------------

typedef __attribute__((ext_vector_type(16))) _Float16 v16h;
typedef __attribute__((ext_vector_type(8)))  float    v8f;

#define ALPHA   0.9f
#define BETA    0.85f
#define SSTEPS  127            // SIM_TIME - 1
#define BROWS   16384
#define INF     128
#define K0      256            // 2*IN
#define H       512

// ---------------- input split + f16 convert: x = [relu(in), relu(-in)] ------
__global__ void k_prep_x(const float* __restrict__ in, _Float16* __restrict__ x) {
  int g = blockIdx.x * blockDim.x + threadIdx.x;      // B*IN threads
  int b = g / INF, i = g % INF;
  float v = in[g];
  x[b * K0 + i]        = (_Float16)fmaxf(v, 0.0f);
  x[b * K0 + INF + i]  = (_Float16)fmaxf(-v, 0.0f);
}

// ---------------- transpose weight to [N][K] f16 (K contiguous) -------------
__global__ void k_wt(const float* __restrict__ w, _Float16* __restrict__ wt,
                     int K, int N) {
  int g = blockIdx.x * blockDim.x + threadIdx.x;      // K*N threads
  int k = g / N, n = g % N;
  wt[n * K + k] = (_Float16)w[g];
}

// ---------------- fragment loaders ------------------------------------------
// A (16x32 f16, row-major source). Per ISA 7.12.2: VGPR v, lanes<16 K=2v,2v+1
// (v<4) / 16+2(v-4).. (v>=4); lanes>=16 add +8 to the K base.
__device__ inline v16h load_a_rowmajor(const _Float16* __restrict__ A, int lda,
                                       int rowbase, int kb, int lane) {
  const _Float16* p = A + (size_t)(rowbase + (lane & 15)) * lda + kb * 32;
  int base = (lane & 16) ? 8 : 0;
  unsigned int u[8];
#pragma unroll
  for (int v = 0; v < 8; ++v) {
    int k = (v < 4) ? (base + 2 * v) : (16 + base + 2 * (v - 4));
    u[v] = *(const unsigned int*)(p + k);             // packed f16 pair
  }
  v16h r; __builtin_memcpy(&r, u, 32);
  return r;
}

// B (32x16 f16) from K-contiguous Wt[N][K]: lane<16 -> col n, K 0..7,16..23;
// lane>=16 -> col n, K 8..15,24..31 (two 16B loads).
__device__ inline v16h load_b_kmajor(const _Float16* __restrict__ Wt, int ldk,
                                     int n0, int kb, int lane) {
  const _Float16* p = Wt + (size_t)(n0 + (lane & 15)) * ldk + kb * 32 +
                      ((lane & 16) ? 8 : 0);
  uint4 u[2];
  u[0] = *(const uint4*)p;
  u[1] = *(const uint4*)(p + 16);
  v16h r; __builtin_memcpy(&r, u, 32);
  return r;
}

// bitmask -> f16 {0,1} A-fragment, built entirely in VALU
__device__ inline v16h mask_to_afrag(unsigned int mask, int base) {
  unsigned int u[8];
#pragma unroll
  for (int v = 0; v < 8; ++v) {
    int k = (v < 4) ? (base + 2 * v) : (16 + base + 2 * (v - 4));
    unsigned int lo = (mask >> k) & 1u;
    unsigned int hi = (mask >> (k + 1)) & 1u;
    u[v] = lo * 0x3C00u + hi * 0x3C000000u;           // packed f16 1.0 pair
  }
  v16h r; __builtin_memcpy(&r, u, 32);
  return r;
}

// ---------------- generic f16 WMMA GEMM: C(f32)[MxN] = A[MxK] * Wt^T --------
__global__ __launch_bounds__(256)
void k_gemm_f16(const _Float16* __restrict__ A, const _Float16* __restrict__ Wt,
                float* __restrict__ C, int K, int N) {
  int lane = threadIdx.x & 31;
  int wave = (blockIdx.x * blockDim.x + threadIdx.x) >> 5;
  int ntiles = N >> 4;
  int tm = wave / ntiles, tn = wave % ntiles;
  int rb = tm * 16, nb = tn * 16;
  v8f acc = {};
  int nkb = K >> 5;
  for (int kb = 0; kb < nkb; ++kb) {
    v16h a = load_a_rowmajor(A, K, rb, kb, lane);
    v16h b = load_b_kmajor(Wt, K, nb, kb, lane);
    acc = __builtin_amdgcn_wmma_f32_16x16x32_f16(false, a, false, b,
                                                 (short)0, acc, false, false);
  }
  int roff = (lane & 16) ? 8 : 0;                     // D: VGPR v -> M=v(+8)
#pragma unroll
  for (int v = 0; v < 8; ++v)
    C[(size_t)(rb + roff + v) * N + nb + (lane & 15)] = acc[v];
}

// ---------------- layer-0 membrane recurrence -> packed spike bits ----------
// spk layout: [t][kb][b] u32, bit j = spike of hidden unit kb*32+j
__global__ __launch_bounds__(256)
void k_spk0(const float* __restrict__ h0, unsigned int* __restrict__ spk) {
  int kb = blockIdx.x >> 6;                           // 16 kb-slices
  int b  = ((blockIdx.x & 63) << 8) | threadIdx.x;    // 64 blocks * 256 = B
  float h[32], m[32];
#pragma unroll
  for (int j = 0; j < 32; ++j) { h[j] = h0[(size_t)b * H + kb * 32 + j]; m[j] = 0.f; }
  for (int t = 0; t < SSTEPS; ++t) {
    unsigned int w = 0;
#pragma unroll
    for (int j = 0; j < 32; ++j) {
      m[j] = BETA * m[j] + h[j];                      // layer0: syn == h0
      if (m[j] > 1.0f) { w |= (1u << j); m[j] = 0.f; }
    }
    spk[(size_t)(t * 16 + kb) * BROWS + b] = w;
  }
}

// ---------------- fused layer-1 time loop + weighted spike sum z ------------
// WG = 256 thr = 8 waves: 2 row-groups x 4 col-groups -> 64 rows x 512 cols.
// Per wave: 2 M-tiles (32 rows) x 128 cols; each B fragment feeds 2 WMMAs.
// s1 is pre-scaled by ALPHA each step, then WMMA accumulates h into it via
// the C operand (no separate h accumulator -> ~128 fewer live VGPRs).
__global__ __launch_bounds__(256)
void k_layer1(const unsigned int* __restrict__ spk,
              const _Float16* __restrict__ W1t, _Float16* __restrict__ z16) {
  int lane = threadIdx.x & 31;
  int wave = threadIdx.x >> 5;                        // 0..7
  int rb = blockIdx.x * 64 + (wave >> 2) * 32;        // 2 row-groups of 32
  int nb = (wave & 3) * 128;
  int base = (lane & 16) ? 8 : 0;

  v8f zf = {};
  v8f s1[2][8], m1[2][8], z[2][8];
#pragma unroll
  for (int mt = 0; mt < 2; ++mt)
#pragma unroll
    for (int i = 0; i < 8; ++i) { s1[mt][i] = zf; m1[mt][i] = zf; z[mt][i] = zf; }

  for (int t = 0; t < SSTEPS; ++t) {
    // s1 *= ALPHA, then accumulate h_t into s1 via WMMA C operand
#pragma unroll
    for (int mt = 0; mt < 2; ++mt)
#pragma unroll
      for (int nt = 0; nt < 8; ++nt)
#pragma unroll
        for (int v = 0; v < 8; ++v) s1[mt][nt][v] *= ALPHA;

    for (int kb = 0; kb < 16; ++kb) {
      const unsigned int* sp = spk + (size_t)(t * 16 + kb) * BROWS + rb;
      v16h a0 = mask_to_afrag(sp[lane & 15], base);
      v16h a1 = mask_to_afrag(sp[16 + (lane & 15)], base);
#pragma unroll
      for (int nt = 0; nt < 8; ++nt) {
        v16h b = load_b_kmajor(W1t, H, nb + nt * 16, kb, lane);
        s1[0][nt] = __builtin_amdgcn_wmma_f32_16x16x32_f16(false, a0, false, b,
                                                           (short)0, s1[0][nt], false, false);
        s1[1][nt] = __builtin_amdgcn_wmma_f32_16x16x32_f16(false, a1, false, b,
                                                           (short)0, s1[1][nt], false, false);
      }
    }

    // closed-form layer-2 collapse weight: c_t = (a^(127-t)-b^(127-t))/(a-b)
    float e  = (float)(SSTEPS - t);
    float ct = (powf(ALPHA, e) - powf(BETA, e)) * (1.0f / (ALPHA - BETA));

#pragma unroll
    for (int mt = 0; mt < 2; ++mt)
#pragma unroll
      for (int nt = 0; nt < 8; ++nt)
#pragma unroll
        for (int v = 0; v < 8; ++v) {
          float m = BETA * m1[mt][nt][v] + s1[mt][nt][v];
          bool sp = m > 1.0f;
          z[mt][nt][v] += sp ? ct : 0.0f;
          m1[mt][nt][v] = sp ? 0.0f : m;
        }
  }

  // write weighted spike sum as f16 (A-operand of the final GEMM)
#pragma unroll
  for (int mt = 0; mt < 2; ++mt)
#pragma unroll
    for (int nt = 0; nt < 8; ++nt)
#pragma unroll
      for (int v = 0; v < 8; ++v)
        z16[(size_t)(rb + mt * 16 + base + v) * H + nb + nt * 16 + (lane & 15)] =
            (_Float16)z[mt][nt][v];
}

// ---------------------------------------------------------------------------
extern "C" void kernel_launch(void* const* d_in, const int* in_sizes, int n_in,
                              void* d_out, int out_size, void* d_ws, size_t ws_size,
                              hipStream_t stream) {
  (void)in_sizes; (void)n_in; (void)out_size; (void)ws_size;
  const float* inp = (const float*)d_in[0];
  const float* W0  = (const float*)d_in[1];
  const float* W1  = (const float*)d_in[2];
  const float* W2  = (const float*)d_in[3];
  float* out = (float*)d_out;

  // workspace carve (256B aligned); total ~193 MB
  char* p = (char*)d_ws;
  auto carve = [&](size_t bytes) { char* r = p; p += (bytes + 255) & ~(size_t)255; return (void*)r; };
  _Float16*     x16  = (_Float16*)    carve((size_t)BROWS * K0 * 2);
  _Float16*     w0t  = (_Float16*)    carve((size_t)H * K0 * 2);
  _Float16*     w1t  = (_Float16*)    carve((size_t)H * H * 2);
  _Float16*     w2t  = (_Float16*)    carve((size_t)H * H * 2);
  float*        h0   = (float*)       carve((size_t)BROWS * H * 4);
  unsigned int* spk  = (unsigned int*)carve((size_t)SSTEPS * 16 * BROWS * 4);
  _Float16*     z16  = (_Float16*)    carve((size_t)BROWS * H * 2);

  // 1) input split -> f16
  k_prep_x<<<(BROWS * INF) / 256, 256, 0, stream>>>(inp, x16);
  // 2) weights -> f16, K-contiguous
  k_wt<<<(K0 * H) / 256, 256, 0, stream>>>(W0, w0t, K0, H);
  k_wt<<<(H * H) / 256, 256, 0, stream>>>(W1, w1t, H, H);
  k_wt<<<(H * H) / 256, 256, 0, stream>>>(W2, w2t, H, H);
  // 3) h0 = x @ W0 (once; time-invariant)   [16384x256]@[256x512]
  k_gemm_f16<<<(BROWS / 16) * (H / 16) / 8, 256, 0, stream>>>(x16, w0t, h0, K0, H);
  // 4) layer-0 recurrence -> packed spike bits (133 MB, L2-resident)
  k_spk0<<<16 * (BROWS / 256), 256, 0, stream>>>(h0, spk);
  // 5) fused layer-1 time loop (127 WMMA GEMMs) + weighted z accumulation
  k_layer1<<<BROWS / 64, 256, 0, stream>>>(spk, w1t, z16);
  // 6) out = z @ W2 (layer-2 linear collapse)  [16384x512]@[512x512]
  k_gemm_f16<<<(BROWS / 16) * (H / 16) / 8, 256, 0, stream>>>(z16, w2t, out, H, H);
}
